// MeshGraphNetsConv_31825707663674
// MI455X (gfx1250) — matrix operand order repaired and checked by reading, not compiled, not run on hardware
//
#include <hip/hip_runtime.h>
#include <hip/hip_bf16.h>

#define N_NODES_ 50000
#define N_EDGES_ 800000

typedef __attribute__((ext_vector_type(16))) __bf16 bf16x16;
typedef __attribute__((ext_vector_type(8)))  __bf16 bf16x8;
typedef __attribute__((ext_vector_type(8)))  float  f32x8;

union ABFrag { bf16x16 v; bf16x8 h[2]; };

__device__ __forceinline__ __bf16 f2bf(float f) {
  union { float f; unsigned u; } x; x.f = f;
  unsigned r = x.u + 0x7FFFu + ((x.u >> 16) & 1u);   // round-to-nearest-even
  unsigned short hs = (unsigned short)(r >> 16);
  return __builtin_bit_cast(__bf16, hs);
}

// One 16xK * Kx16 tile of the GEMM. A is LDS row-major (stride in elements),
// Bsw is the pre-swizzled bf16 weight fragment stream in global memory.
// Accumulator initialized with per-column bias.
template<int KT>
__device__ __forceinline__ f32x8 mm_tile(const __bf16* A, int strideA,
                                         const __bf16* __restrict__ Bsw,
                                         int nt, int lane, float binit) {
  f32x8 c;
  #pragma unroll
  for (int r = 0; r < 8; ++r) c[r] = binit;
  const int m  = lane & 15;
  const int hi = lane >> 4;
  #pragma unroll
  for (int ks = 0; ks < KT; ++ks) {
    ABFrag a, b;
    // 16-bit A layout: lane<16 -> K{0..7,16..23}; lane>=16 -> K{8..15,24..31}
    const __bf16* ap = A + m * strideA + ks * 32 + hi * 8;
    a.h[0] = *(const bf16x8*)(ap);
    a.h[1] = *(const bf16x8*)(ap + 16);
    // B fragments pre-swizzled: 32B contiguous per lane per k-step
    const __bf16* bp = Bsw + ((size_t)((nt * KT + ks) * 32 + lane) << 4);
    b.h[0] = *(const bf16x8*)(bp);
    b.h[1] = *(const bf16x8*)(bp + 8);
    c = __builtin_amdgcn_wmma_f32_16x16x32_bf16(false, a.v, false, b.v,
                                                (short)0, c, false, false);
  }
  return c;
}

// Convert fp32 weight matrix [K x 128] to swizzled bf16 B-fragment stream.
__global__ void swizzle_kernel(const float* __restrict__ src,
                               __bf16* __restrict__ dst, int K) {
  int KT = K >> 5;
  int total = 8 * KT * 32;
  int idx = blockIdx.x * blockDim.x + threadIdx.x;
  if (idx >= total) return;
  int lane  = idx & 31;
  int kstep = (idx >> 5) % KT;
  int nt    = idx / (KT * 32);
  int col   = nt * 16 + (lane & 15);
  int row0  = kstep * 32 + ((lane >> 4) << 3);
  ABFrag f;
  #pragma unroll
  for (int e = 0; e < 16; ++e) f.v[e] = f2bf(src[(size_t)(row0 + (e & 7) + ((e >> 3) << 4)) * 128 + col]);
  bf16x8* d = (bf16x8*)(dst + ((size_t)idx << 4));
  d[0] = f.h[0];
  d[1] = f.h[1];
}

// ---------------- Edge processor ----------------
// Per wave: 16 edges. eh=[x[i]|x[j]|ea] (384) -> MLP(384,128,128) -> LN ->
// residual -> store edge_out + atomic scatter-add into agg (receiver j).
__global__ __launch_bounds__(128) void edge_kernel(
    const float* __restrict__ x, const int* __restrict__ eidx,
    const float* __restrict__ ea,
    const __bf16* __restrict__ w1s, const __bf16* __restrict__ w2s,
    const __bf16* __restrict__ w3s,
    const float* __restrict__ b1, const float* __restrict__ b2,
    const float* __restrict__ b3,
    const float* __restrict__ gma, const float* __restrict__ bta,
    float* __restrict__ eout, float* __restrict__ agg) {
  __shared__ __align__(32) unsigned char smem[4][16384];
  const int lane = threadIdx.x & 31;
  const int wave = threadIdx.x >> 5;
  const int m = lane & 15, hi = lane >> 4;
  __bf16* bufA = (__bf16*)(smem[wave]);            // eh (12KB) -> later h2(4KB)+h3
  __bf16* bufB = (__bf16*)(smem[wave] + 12288);    // h1 (4KB)
  float*  h3   = (float*)(smem[wave] + 4096);      // layer-3 out fp32 (8KB)
  const int* ei = eidx;
  const int* ej = eidx + N_EDGES_;
  const int nTiles = N_EDGES_ / 16;
  for (int tile = blockIdx.x * 4 + wave; tile < nTiles; tile += gridDim.x * 4) {
    // gather + fp32->bf16 staging (float4 coalesced)
    for (int t = lane; t < 16 * 96; t += 32) {
      int em = t / 96, q = t % 96;
      int e = tile * 16 + em;
      const float4* src;
      if (q < 32)      src = (const float4*)(x + (size_t)ei[e] * 128) + q;
      else if (q < 64) src = (const float4*)(x + (size_t)ej[e] * 128) + (q - 32);
      else             src = (const float4*)(ea + (size_t)e * 128) + (q - 64);
      float4 v = *src;
      __bf16* d = bufA + em * 384 + q * 4;
      d[0] = f2bf(v.x); d[1] = f2bf(v.y); d[2] = f2bf(v.z); d[3] = f2bf(v.w);
    }
    // layer 1 (K=384) + SiLU -> bufB
    for (int nt = 0; nt < 8; ++nt) {
      f32x8 c = mm_tile<12>(bufA, 384, w1s, nt, lane, b1[nt * 16 + m]);
      #pragma unroll
      for (int r = 0; r < 8; ++r) {
        float v = c[r]; v = v * (1.f / (1.f + __expf(-v)));
        bufB[(r + 8 * hi) * 128 + nt * 16 + m] = f2bf(v);
      }
    }
    // layer 2 (K=128) + SiLU -> h2 (reuse bufA[0..4KB))
    __bf16* h2 = bufA;
    for (int nt = 0; nt < 8; ++nt) {
      f32x8 c = mm_tile<4>(bufB, 128, w2s, nt, lane, b2[nt * 16 + m]);
      #pragma unroll
      for (int r = 0; r < 8; ++r) {
        float v = c[r]; v = v * (1.f / (1.f + __expf(-v)));
        h2[(r + 8 * hi) * 128 + nt * 16 + m] = f2bf(v);
      }
    }
    // layer 3 (K=128) -> h3 fp32
    for (int nt = 0; nt < 8; ++nt) {
      f32x8 c = mm_tile<4>(h2, 128, w3s, nt, lane, b3[nt * 16 + m]);
      #pragma unroll
      for (int r = 0; r < 8; ++r) h3[(r + 8 * hi) * 128 + nt * 16 + m] = c[r];
    }
    // LayerNorm stats: each 16-lane half owns 8 rows
    float rs[8], rq[8];
    #pragma unroll
    for (int rl = 0; rl < 8; ++rl) {
      float s = 0.f, q = 0.f;
      #pragma unroll
      for (int k = 0; k < 8; ++k) {
        float v = h3[(hi * 8 + rl) * 128 + m + k * 16];
        s += v; q += v * v;
      }
      #pragma unroll
      for (int d = 1; d < 16; d <<= 1) {
        s += __shfl_xor(s, d, 16);
        q += __shfl_xor(q, d, 16);
      }
      rs[rl] = s; rq[rl] = q;
    }
    // normalize + residual + store + scatter-add to receiver j
    for (int i = 0; i < 64; ++i) {
      int rl = i >> 3;
      int col = m + (i & 7) * 16;
      int row = hi * 8 + rl;
      float mu   = rs[rl] * (1.f / 128.f);
      float var  = rq[rl] * (1.f / 128.f) - mu * mu;
      float rstd = rsqrtf(var + 1e-5f);
      int e = tile * 16 + row;
      float hv = h3[row * 128 + col];
      float o = ea[(size_t)e * 128 + col] + (hv - mu) * rstd * gma[col] + bta[col];
      eout[(size_t)e * 128 + col] = o;
      atomicAdd(agg + (size_t)ej[e] * 128 + col, o);
    }
  }
}

// ---------------- Node processor ----------------
// nh=[x|agg] (256) -> MLP(256,128,128) -> LN -> residual. xagg holds agg on
// input and is overwritten in place with x_out.
__global__ __launch_bounds__(128) void node_kernel(
    const float* __restrict__ x, float* xagg,
    const __bf16* __restrict__ w1s, const __bf16* __restrict__ w2s,
    const __bf16* __restrict__ w3s,
    const float* __restrict__ b1, const float* __restrict__ b2,
    const float* __restrict__ b3,
    const float* __restrict__ gma, const float* __restrict__ bta) {
  __shared__ __align__(32) unsigned char smem[4][16384];
  const int lane = threadIdx.x & 31;
  const int wave = threadIdx.x >> 5;
  const int m = lane & 15, hi = lane >> 4;
  __bf16* bufA = (__bf16*)(smem[wave]);
  __bf16* bufB = (__bf16*)(smem[wave] + 12288);
  float*  h3   = (float*)(smem[wave] + 4096);
  const int nTiles = N_NODES_ / 16;
  for (int tile = blockIdx.x * 4 + wave; tile < nTiles; tile += gridDim.x * 4) {
    for (int t = lane; t < 16 * 64; t += 32) {
      int nm = t >> 6, q = t & 63;
      int v = tile * 16 + nm;
      const float4* src = (q < 32)
          ? (const float4*)(x + (size_t)v * 128) + q
          : (const float4*)(xagg + (size_t)v * 128) + (q - 32);
      float4 val = *src;
      __bf16* d = bufA + nm * 256 + q * 4;
      d[0] = f2bf(val.x); d[1] = f2bf(val.y); d[2] = f2bf(val.z); d[3] = f2bf(val.w);
    }
    for (int nt = 0; nt < 8; ++nt) {
      f32x8 c = mm_tile<8>(bufA, 256, w1s, nt, lane, b1[nt * 16 + m]);
      #pragma unroll
      for (int r = 0; r < 8; ++r) {
        float v = c[r]; v = v * (1.f / (1.f + __expf(-v)));
        bufB[(r + 8 * hi) * 128 + nt * 16 + m] = f2bf(v);
      }
    }
    __bf16* h2 = bufA;
    for (int nt = 0; nt < 8; ++nt) {
      f32x8 c = mm_tile<4>(bufB, 128, w2s, nt, lane, b2[nt * 16 + m]);
      #pragma unroll
      for (int r = 0; r < 8; ++r) {
        float v = c[r]; v = v * (1.f / (1.f + __expf(-v)));
        h2[(r + 8 * hi) * 128 + nt * 16 + m] = f2bf(v);
      }
    }
    for (int nt = 0; nt < 8; ++nt) {
      f32x8 c = mm_tile<4>(h2, 128, w3s, nt, lane, b3[nt * 16 + m]);
      #pragma unroll
      for (int r = 0; r < 8; ++r) h3[(r + 8 * hi) * 128 + nt * 16 + m] = c[r];
    }
    float rs[8], rq[8];
    #pragma unroll
    for (int rl = 0; rl < 8; ++rl) {
      float s = 0.f, q = 0.f;
      #pragma unroll
      for (int k = 0; k < 8; ++k) {
        float v = h3[(hi * 8 + rl) * 128 + m + k * 16];
        s += v; q += v * v;
      }
      #pragma unroll
      for (int d = 1; d < 16; d <<= 1) {
        s += __shfl_xor(s, d, 16);
        q += __shfl_xor(q, d, 16);
      }
      rs[rl] = s; rq[rl] = q;
    }
    for (int i = 0; i < 64; ++i) {
      int rl = i >> 3;
      int col = m + (i & 7) * 16;
      int row = hi * 8 + rl;
      float mu   = rs[rl] * (1.f / 128.f);
      float var  = rq[rl] * (1.f / 128.f) - mu * mu;
      float rstd = rsqrtf(var + 1e-5f);
      int v = tile * 16 + row;
      float hv = h3[row * 128 + col];
      float o = x[(size_t)v * 128 + col] + (hv - mu) * rstd * gma[col] + bta[col];
      xagg[(size_t)v * 128 + col] = o;   // overwrite agg with x_out
    }
  }
}

extern "C" void kernel_launch(void* const* d_in, const int* in_sizes, int n_in,
                              void* d_out, int out_size, void* d_ws, size_t ws_size,
                              hipStream_t stream) {
  const float* x    = (const float*)d_in[0];
  const int*   eidx = (const int*)d_in[1];
  const float* ea   = (const float*)d_in[2];
  const float* ew1  = (const float*)d_in[3];
  const float* eb1  = (const float*)d_in[4];
  const float* ew2  = (const float*)d_in[5];
  const float* eb2  = (const float*)d_in[6];
  const float* ew3  = (const float*)d_in[7];
  const float* eb3  = (const float*)d_in[8];
  const float* eg   = (const float*)d_in[9];
  const float* ebt  = (const float*)d_in[10];
  const float* nw1  = (const float*)d_in[11];
  const float* nb1  = (const float*)d_in[12];
  const float* nw2  = (const float*)d_in[13];
  const float* nb2  = (const float*)d_in[14];
  const float* nw3  = (const float*)d_in[15];
  const float* nb3  = (const float*)d_in[16];
  const float* ng   = (const float*)d_in[17];
  const float* nbt  = (const float*)d_in[18];

  float* out      = (float*)d_out;
  float* xout_agg = out;                                   // [N,128] x_out (also agg scratch)
  float* eout     = out + (size_t)N_NODES_ * 128;          // [E,128] edge_out

  __bf16* ws   = (__bf16*)d_ws;
  __bf16* ew1s = ws;                       // 384*128
  __bf16* ew2s = ew1s + 384 * 128;         // 128*128
  __bf16* ew3s = ew2s + 128 * 128;
  __bf16* nw1s = ew3s + 128 * 128;         // 256*128
  __bf16* nw2s = nw1s + 256 * 128;
  __bf16* nw3s = nw2s + 128 * 128;

  // zero agg region (x_out area of d_out) before scatter-add
  hipMemsetAsync(xout_agg, 0, (size_t)N_NODES_ * 128 * sizeof(float), stream);

  // pre-swizzle weights to bf16 B-fragment streams
  auto swz = [&](const float* s, __bf16* d, int K) {
    int total = 8 * (K >> 5) * 32;
    swizzle_kernel<<<(total + 127) / 128, 128, 0, stream>>>(s, d, K);
  };
  swz(ew1, ew1s, 384);
  swz(ew2, ew2s, 128);
  swz(ew3, ew3s, 128);
  swz(nw1, nw1s, 256);
  swz(nw2, nw2s, 128);
  swz(nw3, nw3s, 128);

  edge_kernel<<<1250, 128, 0, stream>>>(x, eidx, ea, ew1s, ew2s, ew3s,
                                        eb1, eb2, eb3, eg, ebt, eout, xout_agg);
  node_kernel<<<400, 128, 0, stream>>>(x, xout_agg, nw1s, nw2s, nw3s,
                                       nb1, nb2, nb3, ng, nbt);
}
